// NoiseAwareAttention_12876311953830
// MI455X (gfx1250) — matrix operand order repaired and checked by reading, not compiled, not run on hardware
//
#include <hip/hip_runtime.h>

// ---------------------------------------------------------------------------
// Noise-aware windowed attention, fully fused per batch for MI455X (gfx1250).
//
// Roofline: ~170 GFLOP total vs ~0.4 GB of unavoidable HBM traffic when fused
// (x in + out; weights stay in the 192MB L2).  bf16 WMMA (16x16x32) with f32
// accumulation is the throughput path: the f32 WMMA shape is only 16x16x4.
// Fusing q/k/v + probs + per-head outputs into the 320KB WGP LDS avoids
// ~1.2 GB of intermediate HBM traffic a multi-kernel pipeline would pay.
//
// Round-2 change: weights are transposed+converted to bf16 once per launch
// (1.2MB workspace), so every WMMA B-fragment is two contiguous
// global_load_b128s from L2 instead of 16 strided b32 loads + 16 VALU
// conversions inside the GEMM K-loops.
// ---------------------------------------------------------------------------

namespace {
constexpr int kN      = 64;    // tokens per window
constexpr int kC      = 384;   // channels
constexpr int kH      = 12;    // heads
constexpr int kD      = 32;    // head dim
constexpr int kC3     = 1152;  // 3*C
constexpr int kHid    = 64;    // trunk hidden
constexpr int kTemb   = 384;
constexpr int kThreads = 384;  // 12 wave32 waves -> one head per wave
constexpr int kWaves   = 12;

// Dynamic-LDS byte offsets (total 316,544 B < 320KB WGP LDS).
constexpr int OFF_BUFA = 0;       // xs 64x384 bf16 (49KB), later probs 12x64x64 bf16 (96KB)
constexpr int OFF_QK   = 98304;   // q|k rows: 64 x 768 bf16 (q cols 0..383, k cols 384..767)
constexpr int OFF_VST  = 196608;  // v transposed: 384 x 64 bf16 (contiguous B-fragments)
constexpr int OFF_OUTS = 245760;  // attention output 64 x 384 bf16
constexpr int OFF_TMOD = 294912;  // temb@qkvt_w + qkv_b : 1152 f32
constexpr int OFF_RPBI = 299520;  // rpb_index copy : 4096 i32
constexpr int OFF_HID1 = 315904;  // 64 f32
constexpr int OFF_HID2 = 316160;  // 64 f32
constexpr int OFF_NB   = 316416;  // 12 f32 (per-head noise bias)
constexpr int OFF_IG   = 316480;  // 1 f32  (1/(1+gate))
constexpr int kSmemBytes = 316544;

// Workspace layout (bytes): bf16 transposed weights.
constexpr size_t WS_QKVWT = 0;                         // [kC3][kC] bf16 = 884,736 B
constexpr size_t WS_PROJT = (size_t)kC3 * kC * 2;      // [kC][kC]  bf16 = 294,912 B
}  // namespace

typedef __attribute__((ext_vector_type(16))) __bf16 bf16x16;
typedef __attribute__((ext_vector_type(8)))  float  f32x8;

union FragU {
  bf16x16        b;
  unsigned short s[16];
};

__device__ __forceinline__ unsigned short f2bf(float x) {
  // Native f32->bf16 (RNE) via the hardware convert.
  union { __bf16 b; unsigned short s; } v;
  v.b = (__bf16)x;
  return v.s;
}

__device__ __forceinline__ f32x8 wmma_bf16(bf16x16 a, bf16x16 b, f32x8 c) {
  // D(16x16,f32) = A(16x32,bf16) x B(32x16,bf16) + C
  return __builtin_amdgcn_wmma_f32_16x16x32_bf16(false, a, false, b,
                                                 (short)0, c, false, false);
}

// A fragment (16x32 bf16) from a row-major bf16 matrix, row stride rs elements.
// ISA layout: lanes 0..15 hold row M=lane, K {kbase+0..7, kbase+16..23};
// lanes 16..31 hold K {kbase+8..15, kbase+24..31}.  Works for LDS or global.
__device__ __forceinline__ bf16x16 frag_a_rm(const unsigned short* src, int rs,
                                             int row0, int kbase, int lane) {
  const int hf = lane >> 4;
  const int r  = lane & 15;
  const unsigned short* p = src + (size_t)(row0 + r) * rs + kbase + hf * 8;
  uint4 lo = *(const uint4*)(p);        // 8 bf16
  uint4 hi = *(const uint4*)(p + 16);   // 8 bf16 at k+16
  FragU f;
  *(uint4*)&f.s[0] = lo;
  *(uint4*)&f.s[8] = hi;
  return f.b;
}

// B fragment (32x16 bf16) where element (k, n) lives at src[(col0+n)*rs + k]
// (i.e. the matrix is stored N-major / transposed).  Two contiguous 16B loads
// per lane.  Works for LDS or global (bf16 transposed weights in workspace).
__device__ __forceinline__ bf16x16 frag_b_nm(const unsigned short* src, int rs,
                                             int col0, int kbase, int lane) {
  const int hf = lane >> 4;
  const int n  = lane & 15;
  const unsigned short* p = src + (size_t)(col0 + n) * rs + kbase + hf * 8;
  uint4 lo = *(const uint4*)(p);        // 8 bf16
  uint4 hi = *(const uint4*)(p + 16);   // 8 bf16 at k+16
  FragU f;
  *(uint4*)&f.s[0] = lo;
  *(uint4*)&f.s[8] = hi;
  return f.b;
}

// -------- Pre-pass: W[K][N] f32 (global) -> Wt[N][K] bf16 (workspace) -------
__global__ void __launch_bounds__(256)
transpose_bf16_kernel(const float* __restrict__ W, unsigned short* __restrict__ Wt,
                      int K, int N) {
  const int idx = blockIdx.x * 256 + threadIdx.x;
  if (idx >= K * N) return;
  const int k = idx / N;          // consecutive idx -> consecutive n: coalesced read
  const int n = idx - k * N;
  Wt[(size_t)n * K + k] = f2bf(W[idx]);
}

// ---------------------------- Fused main kernel ----------------------------
__global__ void __launch_bounds__(kThreads)
naa_fused_kernel(const float* __restrict__ x, const float* __restrict__ temb,
                 const float* __restrict__ sigma,
                 const unsigned short* __restrict__ qkv_wt,   // [kC3][kC] bf16 (transposed)
                 const float* __restrict__ qkv_bias,
                 const float* __restrict__ qkvt_w,
                 const float* __restrict__ trunk_w1, const float* __restrict__ trunk_b1,
                 const float* __restrict__ trunk_w2, const float* __restrict__ trunk_b2,
                 const float* __restrict__ gate_w, const float* __restrict__ gate_b,
                 const float* __restrict__ bias_w, const float* __restrict__ bias_b,
                 const unsigned short* __restrict__ proj_wt,  // [kC][kC] bf16 (transposed)
                 const float* __restrict__ proj_b,
                 const float* __restrict__ rpb_table, const int* __restrict__ rpb_index,
                 float* __restrict__ out) {
  extern __shared__ char smem[];
  unsigned short* xs   = (unsigned short*)(smem + OFF_BUFA);
  unsigned short* prob = (unsigned short*)(smem + OFF_BUFA);   // aliases xs (dead by then)
  unsigned short* qk   = (unsigned short*)(smem + OFF_QK);
  unsigned short* vst  = (unsigned short*)(smem + OFF_VST);
  unsigned short* outs = (unsigned short*)(smem + OFF_OUTS);
  float* tmod = (float*)(smem + OFF_TMOD);
  int*   rpbi = (int*)(smem + OFF_RPBI);
  float* hid1 = (float*)(smem + OFF_HID1);
  float* hid2 = (float*)(smem + OFF_HID2);
  float* nb_l = (float*)(smem + OFF_NB);
  float* ig_l = (float*)(smem + OFF_IG);

  const int tid  = threadIdx.x;
  const int lane = tid & 31;
  const int wave = tid >> 5;
  const int bb   = blockIdx.x;
  const f32x8 zero8 = {0.f, 0.f, 0.f, 0.f, 0.f, 0.f, 0.f, 0.f};

  // -------- Phase 1: stage x (f32 -> bf16 LDS), temb modulation, rpb idx, MLP1
  {
    const float4* xg4 = (const float4*)(x + (size_t)bb * kN * kC);
    for (int i = tid; i < kN * kC / 4; i += kThreads) {
      float4 v4 = xg4[i];
      const int base = i * 4;
      xs[base + 0] = f2bf(v4.x);
      xs[base + 1] = f2bf(v4.y);
      xs[base + 2] = f2bf(v4.z);
      xs[base + 3] = f2bf(v4.w);
    }
    const float* tb = temb + (size_t)bb * kTemb;
    for (int c = tid; c < kC3; c += kThreads) {       // fold qkv_b in here
      float acc = qkv_bias[c];
      for (int t = 0; t < kTemb; ++t) acc += tb[t] * qkvt_w[(size_t)t * kC3 + c];
      tmod[c] = acc;
    }
    for (int i = tid; i < kN * kN; i += kThreads) rpbi[i] = rpb_index[i];
    if (tid < kHid) {
      const float ls = logf(fmaxf(sigma[bb], 1e-6f));
      const float z  = ls * trunk_w1[tid] + trunk_b1[tid];
      hid1[tid] = z / (1.f + __expf(-z));             // silu
    }
  }
  __syncthreads();

  // -------- Phase 2: trunk layer 2, then QKV GEMM (64x384 @ 384x1152)
  if (tid < kHid) {
    float z = trunk_b2[tid];
    for (int i = 0; i < kHid; ++i) z += hid1[i] * trunk_w2[i * kHid + tid];
    hid2[tid] = z / (1.f + __expf(-z));
  }

  for (int t = wave; t < (kN / 16) * (kC3 / 16); t += kWaves) {   // 288 tiles
    const int tr   = t & 3;
    const int tc   = t >> 2;
    const int col0 = tc * 16;
    f32x8 acc = zero8;
#pragma unroll
    for (int ks = 0; ks < kC / 32; ++ks) {
      __builtin_prefetch(qkv_wt + (size_t)col0 * kC + (ks + 1) * 32, 0, 1);
      bf16x16 a = frag_a_rm(xs, kC, tr * 16, ks * 32, lane);
      bf16x16 b = frag_b_nm(qkv_wt, kC, col0, ks * 32, lane);
      acc = wmma_bf16(a, b, acc);
    }
    const int hf = lane >> 4;
    const int cn = col0 + (lane & 15);
    const float add = tmod[cn];
#pragma unroll
    for (int r = 0; r < 8; ++r) {
      const int row = tr * 16 + hf * 8 + r;           // C/D layout: M = r(+8), N = lane%16
      const float v = acc[r] + add;
      if (cn < 2 * kC) qk[row * (2 * kC) + cn] = f2bf(v);            // q | k row-major
      else             vst[(cn - 2 * kC) * kN + row] = f2bf(v);      // v transposed
    }
  }
  __syncthreads();

  // -------- Phase 2b: noise gate and per-head bias from hid2
  if (tid < kH) {
    float z = bias_b[tid];
    for (int i = 0; i < kHid; ++i) z += hid2[i] * bias_w[i * kH + tid];
    nb_l[tid] = z;
  } else if (tid == kH) {
    float z = gate_b[0];
    for (int i = 0; i < kHid; ++i) z += hid2[i] * gate_w[i];
    ig_l[0] = 1.f / (1.f + (1.f / (1.f + __expf(-z))));   // 1/(1+sigmoid(z))
  }
  __syncthreads();

  // -------- Phase 3: attention, one head per wave
  {
    const int h = wave;
    const unsigned short* qh = qk + h * kD;           // stride 768 row-major
    const unsigned short* kh = qk + kC + h * kD;
    const float ig    = ig_l[0];
    const float nb    = nb_l[h];
    const float scale = 0.17677669529663687f;         // 1/sqrt(32)
    const int hf  = lane >> 4;
    const int n15 = lane & 15;

    // scores S = q @ k^T : 4x4 tiles of 16x16, single K=32 chunk (D=32)
    f32x8 S[4][4];
#pragma unroll
    for (int tr = 0; tr < 4; ++tr) {
      bf16x16 a = frag_a_rm(qh, 2 * kC, tr * 16, 0, lane);
#pragma unroll
      for (int tc = 0; tc < 4; ++tc) {
        bf16x16 b = frag_b_nm(kh, 2 * kC, tc * 16, 0, lane);
        S[tr][tc] = wmma_bf16(a, b, zero8);
      }
    }

    // epilogue: scale, relative-position bias, noise gate/bias
#pragma unroll
    for (int tr = 0; tr < 4; ++tr)
#pragma unroll
      for (int tc = 0; tc < 4; ++tc)
#pragma unroll
        for (int r = 0; r < 8; ++r) {
          const int row = tr * 16 + hf * 8 + r;
          const int col = tc * 16 + n15;
          const float rv = rpb_table[rpbi[row * kN + col] * kH + h];
          S[tr][tc][r] = (S[tr][tc][r] * scale + rv) * ig + nb;
        }

    // row softmax: each row lives in one 16-lane half at a fixed vgpr slot,
    // so xor-shuffles with mask<16 reduce exactly that row.
#pragma unroll
    for (int tr = 0; tr < 4; ++tr)
#pragma unroll
      for (int r = 0; r < 8; ++r) {
        float mx = S[tr][0][r];
#pragma unroll
        for (int tc = 1; tc < 4; ++tc) mx = fmaxf(mx, S[tr][tc][r]);
#pragma unroll
        for (int m = 1; m < 16; m <<= 1) mx = fmaxf(mx, __shfl_xor(mx, m, 32));
        float sum = 0.f;
#pragma unroll
        for (int tc = 0; tc < 4; ++tc) {
          const float e = __expf(S[tr][tc][r] - mx);
          S[tr][tc][r] = e;
          sum += e;
        }
#pragma unroll
        for (int m = 1; m < 16; m <<= 1) sum += __shfl_xor(sum, m, 32);
        const float inv = 1.f / sum;
#pragma unroll
        for (int tc = 0; tc < 4; ++tc) S[tr][tc][r] *= inv;
      }

    // re-layout probs through per-wave LDS scratch into A-fragment order
    unsigned short* pw = prob + wave * (kN * kN);
#pragma unroll
    for (int tr = 0; tr < 4; ++tr)
#pragma unroll
      for (int tc = 0; tc < 4; ++tc)
#pragma unroll
        for (int r = 0; r < 8; ++r) {
          const int row = tr * 16 + hf * 8 + r;
          const int col = tc * 16 + n15;
          pw[row * kN + col] = f2bf(S[tr][tc][r]);
        }
    __asm__ volatile("s_wait_dscnt 0" ::: "memory");  // same-wave LDS RAW

    // out_h = P(64x64) @ v_h(64x32); v stored transposed -> contiguous B frags
    f32x8 O[4][2];
#pragma unroll
    for (int tr = 0; tr < 4; ++tr)
#pragma unroll
      for (int tc = 0; tc < 2; ++tc) O[tr][tc] = zero8;
    const unsigned short* vh = vst + h * kD * kN;
#pragma unroll
    for (int kt = 0; kt < 2; ++kt)
#pragma unroll
      for (int tr = 0; tr < 4; ++tr) {
        bf16x16 a = frag_a_rm(pw, kN, tr * 16, kt * 32, lane);
#pragma unroll
        for (int tc = 0; tc < 2; ++tc) {
          bf16x16 b = frag_b_nm(vh, kN, tc * 16, kt * 32, lane);
          O[tr][tc] = wmma_bf16(a, b, O[tr][tc]);
        }
      }

#pragma unroll
    for (int tr = 0; tr < 4; ++tr)
#pragma unroll
      for (int tc = 0; tc < 2; ++tc)
#pragma unroll
        for (int r = 0; r < 8; ++r) {
          const int row = tr * 16 + hf * 8 + r;
          const int col = h * kD + tc * 16 + n15;
          outs[row * kC + col] = f2bf(O[tr][tc][r]);
        }
  }
  __syncthreads();

  // -------- Phase 4: output projection (64x384 @ 384x384) + bias, f32 out
  float* og = out + (size_t)bb * kN * kC;
  for (int t = wave; t < (kN / 16) * (kC / 16); t += kWaves) {   // 96 tiles
    const int tr   = t & 3;
    const int tc   = t >> 2;
    const int col0 = tc * 16;
    f32x8 acc = zero8;
#pragma unroll
    for (int ks = 0; ks < kC / 32; ++ks) {
      __builtin_prefetch(proj_wt + (size_t)col0 * kC + (ks + 1) * 32, 0, 1);
      bf16x16 a = frag_a_rm(outs, kC, tr * 16, ks * 32, lane);
      bf16x16 b = frag_b_nm(proj_wt, kC, col0, ks * 32, lane);
      acc = wmma_bf16(a, b, acc);
    }
    const int hf = lane >> 4;
    const int cn = col0 + (lane & 15);
    const float pb = proj_b[cn];
#pragma unroll
    for (int r = 0; r < 8; ++r) {
      const int row = tr * 16 + hf * 8 + r;
      og[row * kC + cn] = acc[r] + pb;
    }
  }
}

extern "C" void kernel_launch(void* const* d_in, const int* in_sizes, int n_in,
                              void* d_out, int out_size, void* d_ws, size_t ws_size,
                              hipStream_t stream) {
  (void)n_in; (void)out_size; (void)ws_size;
  const float* x        = (const float*)d_in[0];
  const float* temb     = (const float*)d_in[1];
  const float* sigma    = (const float*)d_in[2];
  const float* qkv_w    = (const float*)d_in[3];
  const float* qkv_b    = (const float*)d_in[4];
  const float* qkvt_w   = (const float*)d_in[5];
  const float* trunk_w1 = (const float*)d_in[6];
  const float* trunk_b1 = (const float*)d_in[7];
  const float* trunk_w2 = (const float*)d_in[8];
  const float* trunk_b2 = (const float*)d_in[9];
  const float* gate_w   = (const float*)d_in[10];
  const float* gate_b   = (const float*)d_in[11];
  const float* bias_w   = (const float*)d_in[12];
  const float* bias_b   = (const float*)d_in[13];
  const float* proj_w   = (const float*)d_in[14];
  const float* proj_b   = (const float*)d_in[15];
  const float* rpb_tab  = (const float*)d_in[16];
  const int*   rpb_idx  = (const int*)d_in[17];
  float* out = (float*)d_out;

  unsigned short* qkv_wt  = (unsigned short*)((char*)d_ws + WS_QKVWT);
  unsigned short* proj_wt = (unsigned short*)((char*)d_ws + WS_PROJT);

  // Pre-pass: transpose + convert weights to bf16 (runs once per launch,
  // ~2.4MB of reads; makes every WMMA B-fragment two contiguous b128 loads).
  {
    const int n_qkv = kC * kC3;
    transpose_bf16_kernel<<<dim3((n_qkv + 255) / 256), dim3(256), 0, stream>>>(
        qkv_w, qkv_wt, kC, kC3);
    const int n_proj = kC * kC;
    transpose_bf16_kernel<<<dim3((n_proj + 255) / 256), dim3(256), 0, stream>>>(
        proj_w, proj_wt, kC, kC);
  }

  const int B = in_sizes[2];  // sigma has one element per batch
  naa_fused_kernel<<<dim3(B), dim3(kThreads), kSmemBytes, stream>>>(
      x, temb, sigma, qkv_wt, qkv_b, qkvt_w, trunk_w1, trunk_b1, trunk_w2,
      trunk_b2, gate_w, gate_b, bias_w, bias_b, proj_wt, proj_b, rpb_tab,
      rpb_idx, out);
}
